// QuantizedLinear_6451040878933
// MI455X (gfx1250) — 8-bit matrix operand order repaired and checked by reading, compiled, not run on hardware
//
#include <hip/hip_runtime.h>


typedef int v8i __attribute__((ext_vector_type(8)));
typedef int v4i __attribute__((ext_vector_type(4)));
typedef int v2i __attribute__((ext_vector_type(2)));

typedef __attribute__((__vector_size__(4 * sizeof(int)))) int vs4i;

#define TILE_M 128
#define TILE_N 256
#define TILE_K 64

__device__ __forceinline__ int clamp_i8(int v) {
    v = v < -128 ? -128 : v;
    return v > 127 ? 127 : v;
}

__device__ __forceinline__ void async_ld16(void* lds_dst, const void* gsrc) {
#if __has_builtin(__builtin_amdgcn_global_load_async_to_lds_b128)
    __builtin_amdgcn_global_load_async_to_lds_b128(
        (__attribute__((address_space(1))) vs4i*)(uintptr_t)gsrc,
        (__attribute__((address_space(3))) vs4i*)(uintptr_t)lds_dst,
        0, 0);
#else
    unsigned ldsoff = (unsigned)(uintptr_t)lds_dst;
    asm volatile("global_load_async_to_lds_b128 %0, %1, off"
                 :: "v"(ldsoff), "v"(gsrc) : "memory");
#endif
}

__device__ __forceinline__ void wait_async0() {
#if __has_builtin(__builtin_amdgcn_s_wait_asynccnt)
    __builtin_amdgcn_s_wait_asynccnt(0);
#else
    asm volatile("s_wait_asynccnt 0" ::: "memory");
#endif
}

__global__ __launch_bounds__(256) void quantize_kernel(
    const float* __restrict__ in,
    int* __restrict__ qout,
    const float* __restrict__ scale_p,
    const int* __restrict__ zp_p,
    int total4)
{
    int idx = blockIdx.x * blockDim.x + threadIdx.x;
    if (idx >= total4) return;
    const float inv = 1.0f / scale_p[0];
    const int zp = zp_p[0];
    float4 x = ((const float4*)in)[idx];
    int a = clamp_i8((int)rintf(x.x * inv) + zp);
    int b = clamp_i8((int)rintf(x.y * inv) + zp);
    int c = clamp_i8((int)rintf(x.z * inv) + zp);
    int d = clamp_i8((int)rintf(x.w * inv) + zp);
    qout[idx] = (a & 0xff) | ((b & 0xff) << 8) | ((c & 0xff) << 16) | ((d & 0xff) << 24);
}

__global__ __launch_bounds__(256) void gemm_iu8_kernel(
    const signed char* __restrict__ Aq,
    const signed char* __restrict__ W,
    const int* __restrict__ bias,
    float* __restrict__ out,
    const float* __restrict__ oscale_p,
    const int* __restrict__ ozp_p,
    int K, int OUT)
{
    __shared__ __align__(16) signed char As[2][TILE_M][TILE_K];
    __shared__ __align__(16) signed char Bs[2][TILE_N][TILE_K];

    const int tid    = threadIdx.x;
    const int lane   = tid & 31;
    const int wave   = tid >> 5;
    const int wave_m = wave >> 2;
    const int wave_n = wave & 3;
    const int bm     = blockIdx.y * TILE_M;
    const int bn     = blockIdx.x * TILE_N;

    v8i acc[4][4];
#pragma unroll
    for (int i = 0; i < 4; ++i)
#pragma unroll
        for (int j = 0; j < 4; ++j)
#pragma unroll
            for (int e = 0; e < 8; ++e) acc[i][j][e] = 0;

    const int arow  = tid >> 1;
    const int ahalf = (tid & 1) * 32;
    const signed char* ga = Aq + (size_t)(bm + arow) * K + ahalf;
    const signed char* gb = W  + (size_t)(bn + tid)  * K;

    const int half = lane >> 4;
    const int lm   = lane & 15;

    const int steps = K / TILE_K;

    {
        async_ld16(&As[0][arow][ahalf],      ga);
        async_ld16(&As[0][arow][ahalf + 16], ga + 16);
        async_ld16(&Bs[0][tid][0],  gb);
        async_ld16(&Bs[0][tid][16], gb + 16);
        async_ld16(&Bs[0][tid][32], gb + 32);
        async_ld16(&Bs[0][tid][48], gb + 48);
    }

    for (int s = 0; s < steps; ++s) {
        const int buf = s & 1;
        wait_async0();
        __syncthreads();

        if (s + 1 < steps) {
            const int nk = (s + 1) * TILE_K;
            const signed char* na = ga + nk;
            const signed char* nb = gb + nk;
            const int nb_buf = buf ^ 1;
            async_ld16(&As[nb_buf][arow][ahalf],      na);
            async_ld16(&As[nb_buf][arow][ahalf + 16], na + 16);
            async_ld16(&Bs[nb_buf][tid][0],  nb);
            async_ld16(&Bs[nb_buf][tid][16], nb + 16);
            async_ld16(&Bs[nb_buf][tid][32], nb + 32);
            async_ld16(&Bs[nb_buf][tid][48], nb + 48);
        }

        v8i afrag[4];
#pragma unroll
        for (int ti = 0; ti < 4; ++ti) {
            const signed char* as = &As[buf][wave_m * 64 + ti * 16 + lm][half * 16];
            union { v8i v; v2i d[4]; } ua;
            ua.d[0] = *(const v2i*)(as);
            ua.d[1] = *(const v2i*)(as + 8);
            ua.d[2] = *(const v2i*)(as + 32);
            ua.d[3] = *(const v2i*)(as + 40);
            afrag[ti] = ua.v;
        }

#pragma unroll
        for (int tj = 0; tj < 4; ++tj) {
            const signed char* bs = &Bs[buf][wave_n * 64 + tj * 16 + lm][half * 16];
            union { v8i v; v4i q[2]; } ub;
            ub.q[0] = *(const v4i*)(bs);
            ub.q[1] = *(const v4i*)(bs + 32);
#pragma unroll
            for (int ti = 0; ti < 4; ++ti) {
                acc[ti][tj] = __builtin_amdgcn_wmma_i32_16x16x64_iu8(
                    true, afrag[ti], true, ub.v, acc[ti][tj], false, false);
            }
        }
        __syncthreads();
    }

    const float inv_os = 1.0f / oscale_p[0];
    const int   ozp    = ozp_p[0];
    const int   rbase  = half * 8;
#pragma unroll
    for (int ti = 0; ti < 4; ++ti) {
#pragma unroll
        for (int tj = 0; tj < 4; ++tj) {
            const int gcol = bn + wave_n * 64 + tj * 16 + lm;
            const int bsum = bias[gcol];
            union { v8i v; int i[8]; } uc;
            uc.v = acc[ti][tj];
#pragma unroll
            for (int r = 0; r < 8; ++r) {
                const int grow = bm + wave_m * 64 + ti * 16 + rbase + r;
                const int a32  = uc.i[r] + bsum;
                int q = (int)rintf((float)a32 * inv_os) + ozp;
                q = clamp_i8(q);
                out[(size_t)grow * OUT + gcol] = (float)q;
            }
        }
    }
}

extern "C" void kernel_launch(void* const* d_in, const int* in_sizes, int n_in,
                              void* d_out, int out_size, void* d_ws, size_t ws_size,
                              hipStream_t stream) {
    const float* input     = (const float*)d_in[0];
    const signed char* W   = (const signed char*)d_in[1];
    const int*   bias      = (const int*)d_in[2];
    const float* in_scale  = (const float*)d_in[3];
    const int*   in_zp     = (const int*)d_in[4];
    const float* out_scale = (const float*)d_in[5];
    const int*   out_zp    = (const int*)d_in[6];

    const int OUT   = in_sizes[2];
    const int K     = in_sizes[1] / OUT;
    const int Nrows = in_sizes[0] / K;

    signed char* Aq = (signed char*)d_ws;
    float* out = (float*)d_out;

    const int total4  = (Nrows * K) / 4;
    const int qblocks = (total4 + 255) / 256;
    quantize_kernel<<<qblocks, 256, 0, stream>>>(input, (int*)Aq, in_scale, in_zp, total4);

    dim3 grid(OUT / TILE_N, Nrows / TILE_M);
    gemm_iu8_kernel<<<grid, 256, 0, stream>>>(Aq, W, bias, out, out_scale, out_zp, K, OUT);
}
